// RSSM_6511170420741
// MI455X (gfx1250) — matrix operand order repaired and checked by reading, not compiled, not run on hardware
//
#include <hip/hip_runtime.h>
#include <math.h>

// ---------------------------------------------------------------------------
// RSSM observe() for MI455X (gfx1250): bf16 WMMA GEMMs + fused glue kernels.
// B=512, T=64, STOCH=256, DETER=1024, HIDDEN=1024, EMBED=1536, ACT=32
// ---------------------------------------------------------------------------

#define B_     512
#define T_     64
#define STOCH_ 256
#define DETER_ 1024
#define HIDDEN_ 1024
#define EMBED_ 1536
#define ACT_   32

typedef __bf16  v16bf __attribute__((ext_vector_type(16)));
typedef float   v8f   __attribute__((ext_vector_type(8)));
typedef unsigned uint4v __attribute__((ext_vector_type(4)));

union Frag { uint4v q[2]; v16bf v; };

__device__ __forceinline__ unsigned short f2bf(float f) {
    union { float f; unsigned u; } v; v.f = f;
    unsigned r = v.u + 0x7FFFu + ((v.u >> 16) & 1u);   // round-to-nearest-even
    return (unsigned short)(r >> 16);
}

__device__ __forceinline__ float sigmoidf_(float x) { return 1.f / (1.f + expf(-x)); }

// ---------------------------------------------------------------------------
// Pack fp32 weight matrix [K,N] (row-major) into WMMA B-fragment bf16 layout:
// per (ktile, ntile) of 32x16, per lane l (n = l&15, khalf = (l>>4)*16),
// 8 consecutive dwords hold bf16 pairs (K = khalf+2j, khalf+2j+1).
// Output dword index == flat thread index by construction.
// ---------------------------------------------------------------------------
__global__ void pack_weights(const float* __restrict__ W, unsigned* __restrict__ Wp,
                             int K, int N) {
    size_t idx = (size_t)blockIdx.x * blockDim.x + threadIdx.x;
    size_t total = (size_t)(K >> 5) * (N >> 4) * 256;
    if (idx >= total) return;
    int j    = (int)(idx & 7);
    int lane = (int)((idx >> 3) & 31);
    size_t tile = idx >> 8;
    int ntiles = N >> 4;
    int nt = (int)(tile % ntiles);
    int kt = (int)(tile / ntiles);
    int k = (kt << 5) + ((lane >> 4) << 4) + 2 * j;
    int n = (nt << 4) + (lane & 15);
    unsigned lo = f2bf(W[(size_t)k * N + n]);
    unsigned hi = f2bf(W[(size_t)(k + 1) * N + n]);
    Wp[idx] = lo | (hi << 16);
}

// Zero bf16 activation arenas (A1,A2,A4 contiguous) + f32 deter carry.
__global__ void zero_ws(unsigned short* __restrict__ arena, size_t nU,
                        float* __restrict__ det, size_t nF) {
    size_t i = (size_t)blockIdx.x * blockDim.x + threadIdx.x;
    if (i < nU) arena[i] = 0;
    if (i < nF) det[i] = 0.f;
}

// Per-step prep: shifted action -> A1[:,256:288], embed[t] -> A4[:,1024:2560]
__global__ void prep_step(const float* __restrict__ embed, const float* __restrict__ action,
                          unsigned short* __restrict__ A1, unsigned short* __restrict__ A4,
                          int t) {
    size_t idx = (size_t)blockIdx.x * blockDim.x + threadIdx.x;
    const size_t nAct = (size_t)B_ * ACT_;
    if (idx < nAct) {
        int b = (int)(idx / ACT_), j = (int)(idx % ACT_);
        float a = (t == 0) ? 0.f : action[((size_t)b * T_ + (t - 1)) * ACT_ + j];
        A1[(size_t)b * (STOCH_ + ACT_) + STOCH_ + j] = f2bf(a);
        return;
    }
    idx -= nAct;
    const size_t nEmb = (size_t)B_ * EMBED_;
    if (idx < nEmb) {
        int b = (int)(idx / EMBED_), e = (int)(idx % EMBED_);
        float v = embed[((size_t)b * T_ + t) * EMBED_ + e];
        A4[(size_t)b * (DETER_ + EMBED_) + DETER_ + e] = f2bf(v);
    }
}

// ---------------------------------------------------------------------------
// WMMA GEMM: C[512,N] = A[512,K](bf16, row-major, stride lda) * Wp(packed bf16)
// 4 waves / block; each wave owns NT consecutive 16x16 N-tiles (A fragment
// reused across NT WMMAs). Block tile = 16(M) x 64*NT(N).
// Strength-reduced pointers: arow += 32 elems, bp += ntiles*256 dwords per k.
// MODE 0: y = elu(acc + bias) -> bf16 arena at [row, col_off+col], stride ldob
// MODE 1: y = acc             -> o0[row*N+col]             (GRU pre-LN, no bias)
// MODE 2: y = acc + bias; col<256 -> pm[b,t,col]; else ps[b,t,col-256]=2*sig(y/2)+.1
// MODE 3: y = acc + bias      -> o0[row*N+col]             (posterior raw logits)
// ---------------------------------------------------------------------------
template <int MODE, int NT>
__global__ __launch_bounds__(128)
void gemm_wmma(const unsigned short* __restrict__ A, int lda, int K,
               const unsigned* __restrict__ Wp, int N,
               const float* __restrict__ bias,
               float* __restrict__ o0, float* __restrict__ o1,
               unsigned short* __restrict__ ob, int ldob, int col_off,
               int t) {
    const int lane  = threadIdx.x & 31;
    const int wave  = threadIdx.x >> 5;
    const int m     = lane & 15;
    const int half  = lane >> 4;
    const int nbase = (blockIdx.x * 4 + wave) * (16 * NT);
    const int mbase = blockIdx.y * 16;
    const int ntiles = N >> 4;

    const unsigned short* arow = A + (size_t)(mbase + m) * lda + half * 16;
    const unsigned* bp = Wp + ((size_t)(nbase >> 4) * 32 + lane) * 8;
    const size_t bstep = (size_t)ntiles * 256;      // dwords per k-tile row

    v8f acc[NT];
#pragma unroll
    for (int s = 0; s < NT; ++s) acc[s] = {};

    for (int k = 0; k < K; k += 32) {
        Frag a;
        a.q[0] = *reinterpret_cast<const uint4v*>(arow);        // K = k+half*16 .. +7
        a.q[1] = *reinterpret_cast<const uint4v*>(arow + 8);    // K = k+half*16+8 .. +15
        arow += 32;
        __builtin_prefetch(bp + 4 * bstep, 0, 1);               // global_prefetch_b8
#pragma unroll
        for (int s = 0; s < NT; ++s) {
            Frag b;
            b.q[0] = *reinterpret_cast<const uint4v*>(bp + s * 256);
            b.q[1] = *reinterpret_cast<const uint4v*>(bp + s * 256 + 4);
            acc[s] = __builtin_amdgcn_wmma_f32_16x16x32_bf16(
                false, a.v, false, b.v, (short)0, acc[s], false, false);
        }
        bp += bstep;
    }

#pragma unroll
    for (int s = 0; s < NT; ++s) {
        const int col = nbase + s * 16 + m;
        const float bcol = (MODE == 1) ? 0.f : bias[col];
#pragma unroll
        for (int j = 0; j < 8; ++j) {
            int row = mbase + half * 8 + j;
            float v = acc[s][j];
            if (MODE == 0) {
                v += bcol;
                v = (v > 0.f) ? v : (expf(v) - 1.f);
                ob[(size_t)row * ldob + col_off + col] = f2bf(v);
            } else if (MODE == 1) {
                o0[(size_t)row * N + col] = v;
            } else if (MODE == 2) {
                v += bcol;
                if (col < STOCH_)
                    o0[((size_t)row * T_ + t) * STOCH_ + col] = v;
                else
                    o1[((size_t)row * T_ + t) * STOCH_ + (col - STOCH_)] =
                        2.f * sigmoidf_(0.5f * v) + 0.1f;
            } else {
                v += bcol;
                o0[(size_t)row * N + col] = v;
            }
        }
    }
}

// ---------------------------------------------------------------------------
// LayerNorm over 3072 + GRU gates. One block per batch row, 256 threads.
// Writes new deter to: carry (f32), output tensor, A2[:,1024:2048], A4[:,0:1024].
// ---------------------------------------------------------------------------
__global__ __launch_bounds__(256)
void gru_ln_gate(const float* __restrict__ gru_out,
                 const float* __restrict__ ln_scale, const float* __restrict__ ln_bias,
                 float* __restrict__ deter_carry, float* __restrict__ out_deter,
                 unsigned short* __restrict__ A2, unsigned short* __restrict__ A4,
                 int t) {
    const int b = blockIdx.x, tid = threadIdx.x;
    const float* row = gru_out + (size_t)b * (3 * DETER_);
    __shared__ float red[256];

    float s = 0.f;
    for (int i = tid; i < 3 * DETER_; i += 256) s += row[i];
    red[tid] = s; __syncthreads();
    for (int st = 128; st > 0; st >>= 1) { if (tid < st) red[tid] += red[tid + st]; __syncthreads(); }
    float mean = red[0] * (1.f / (3 * DETER_));
    __syncthreads();

    float vs = 0.f;
    for (int i = tid; i < 3 * DETER_; i += 256) { float d = row[i] - mean; vs += d * d; }
    red[tid] = vs; __syncthreads();
    for (int st = 128; st > 0; st >>= 1) { if (tid < st) red[tid] += red[tid + st]; __syncthreads(); }
    float rstd = rsqrtf(red[0] * (1.f / (3 * DETER_)) + 1e-6f);
    __syncthreads();

    for (int d = tid; d < DETER_; d += 256) {
        float r = (row[d]              - mean) * rstd * ln_scale[d]              + ln_bias[d];
        float c = (row[d + DETER_]     - mean) * rstd * ln_scale[d + DETER_]     + ln_bias[d + DETER_];
        float u = (row[d + 2 * DETER_] - mean) * rstd * ln_scale[d + 2 * DETER_] + ln_bias[d + 2 * DETER_];
        float rg   = sigmoidf_(r);
        float cand = tanhf(rg * c);
        float ug   = sigmoidf_(u - 1.f);   // UPDATE_BIAS = -1
        float old  = deter_carry[(size_t)b * DETER_ + d];
        float nd   = ug * cand + (1.f - ug) * old;
        deter_carry[(size_t)b * DETER_ + d] = nd;
        out_deter[((size_t)b * T_ + t) * DETER_ + d] = nd;
        unsigned short nb = f2bf(nd);
        A2[(size_t)b * (HIDDEN_ + DETER_) + HIDDEN_ + d] = nb;
        A4[(size_t)b * (DETER_ + EMBED_) + d] = nb;
    }
}

// ---------------------------------------------------------------------------
// Posterior finish: qm/qs from raw logits, stoch = qm + qs*noise[t], write
// outputs and stoch (bf16) back into A1 for step t+1.
// ---------------------------------------------------------------------------
__global__ void finish_post(const float* __restrict__ q_raw, const float* __restrict__ noise,
                            float* __restrict__ out_stoch, float* __restrict__ out_qm,
                            float* __restrict__ out_qs, unsigned short* __restrict__ A1,
                            int t) {
    size_t idx = (size_t)blockIdx.x * blockDim.x + threadIdx.x;
    if (idx >= (size_t)B_ * STOCH_) return;
    int b = (int)(idx / STOCH_), c = (int)(idx % STOCH_);
    float qm = q_raw[(size_t)b * (2 * STOCH_) + c];
    float qs = 2.f * sigmoidf_(0.5f * q_raw[(size_t)b * (2 * STOCH_) + STOCH_ + c]) + 0.1f;
    float nz = noise[((size_t)t * B_ + b) * STOCH_ + c];
    float st = qm + qs * nz;
    size_t o = ((size_t)b * T_ + t) * STOCH_ + c;
    out_stoch[o] = st;
    out_qm[o]    = qm;
    out_qs[o]    = qs;
    A1[(size_t)b * (STOCH_ + ACT_) + c] = f2bf(st);
}

// ---------------------------------------------------------------------------
// Host side
// ---------------------------------------------------------------------------
extern "C" void kernel_launch(void* const* d_in, const int* in_sizes, int n_in,
                              void* d_out, int out_size, void* d_ws, size_t ws_size,
                              hipStream_t stream) {
    const float* embed      = (const float*)d_in[0];
    const float* action     = (const float*)d_in[1];
    const float* noise      = (const float*)d_in[2];
    const float* img_in_w   = (const float*)d_in[3];
    const float* img_in_b   = (const float*)d_in[4];
    const float* gru_w      = (const float*)d_in[5];
    const float* ln_scale   = (const float*)d_in[6];
    const float* ln_bias    = (const float*)d_in[7];
    const float* img_out_w  = (const float*)d_in[8];
    const float* img_out_b  = (const float*)d_in[9];
    const float* img_dist_w = (const float*)d_in[10];
    const float* img_dist_b = (const float*)d_in[11];
    const float* obs_out_w  = (const float*)d_in[12];
    const float* obs_out_b  = (const float*)d_in[13];
    const float* obs_dist_w = (const float*)d_in[14];
    const float* obs_dist_b = (const float*)d_in[15];

    // --- workspace layout (bytes; all sizes multiples of 256) ---
    char* ws = (char*)d_ws;
    size_t o = 0;
    auto take = [&](size_t bytes) { char* p = ws + o; o += bytes; return p; };
    unsigned* WpImgIn   = (unsigned*)take((size_t)288  * 1024 * 2);
    unsigned* WpGru     = (unsigned*)take((size_t)2048 * 3072 * 2);
    unsigned* WpImgOut  = (unsigned*)take((size_t)1024 * 1024 * 2);
    unsigned* WpImgDist = (unsigned*)take((size_t)1024 * 512  * 2);
    unsigned* WpObsOut  = (unsigned*)take((size_t)2560 * 1024 * 2);
    unsigned* WpObsDist = (unsigned*)take((size_t)1024 * 512  * 2);
    unsigned short* A1  = (unsigned short*)take((size_t)B_ * (STOCH_ + ACT_)   * 2);
    unsigned short* A2  = (unsigned short*)take((size_t)B_ * (HIDDEN_ + DETER_) * 2);
    unsigned short* A4  = (unsigned short*)take((size_t)B_ * (DETER_ + EMBED_)  * 2);
    unsigned short* Hb  = (unsigned short*)take((size_t)B_ * HIDDEN_ * 2);
    unsigned short* H2b = (unsigned short*)take((size_t)B_ * HIDDEN_ * 2);
    float* gru_out = (float*)take((size_t)B_ * 3 * DETER_ * 4);
    float* deterC  = (float*)take((size_t)B_ * DETER_ * 4);
    float* q_raw   = (float*)take((size_t)B_ * 2 * STOCH_ * 4);

    // --- output layout: (stoch, deter, qm, qs, pm, ps), each [B,T,dim] flat ---
    float* out = (float*)d_out;
    float* out_stoch = out;
    float* out_deter = out_stoch + (size_t)B_ * T_ * STOCH_;
    float* out_qm    = out_deter + (size_t)B_ * T_ * DETER_;
    float* out_qs    = out_qm    + (size_t)B_ * T_ * STOCH_;
    float* out_pm    = out_qs    + (size_t)B_ * T_ * STOCH_;
    float* out_ps    = out_pm    + (size_t)B_ * T_ * STOCH_;

    // --- one-time per launch: pack weights to bf16 fragments, zero carries ---
    auto pack = [&](const float* W, unsigned* Wp, int K, int N) {
        size_t total = (size_t)K * N / 2;
        pack_weights<<<(unsigned)((total + 255) / 256), 256, 0, stream>>>(W, Wp, K, N);
    };
    pack(img_in_w,   WpImgIn,   288,  1024);
    pack(gru_w,      WpGru,     2048, 3072);
    pack(img_out_w,  WpImgOut,  1024, 1024);
    pack(img_dist_w, WpImgDist, 1024, 512);
    pack(obs_out_w,  WpObsOut,  2560, 1024);
    pack(obs_dist_w, WpObsDist, 1024, 512);

    {
        size_t nU = (size_t)B_ * ((STOCH_ + ACT_) + (HIDDEN_ + DETER_) + (DETER_ + EMBED_));
        size_t nF = (size_t)B_ * DETER_;
        size_t mx = nU > nF ? nU : nF;
        zero_ws<<<(unsigned)((mx + 255) / 256), 256, 0, stream>>>(A1, nU, deterC, nF);
    }

    const dim3 blk(128);
    for (int t = 0; t < T_; ++t) {
        {   // shifted action + embed slice -> bf16 arenas
            size_t total = (size_t)B_ * ACT_ + (size_t)B_ * EMBED_;
            prep_step<<<(unsigned)((total + 255) / 256), 256, 0, stream>>>(embed, action, A1, A4, t);
        }
        // x = elu(A1 @ img_in_w + b) -> A2[:,0:1024]
        gemm_wmma<0, 2><<<dim3(HIDDEN_ / 128, B_ / 16), blk, 0, stream>>>(
            A1, STOCH_ + ACT_, STOCH_ + ACT_, WpImgIn, HIDDEN_, img_in_b,
            nullptr, nullptr, A2, HIDDEN_ + DETER_, 0, t);
        // parts_raw = A2 @ gru_w -> gru_out (no bias)
        gemm_wmma<1, 2><<<dim3(3 * DETER_ / 128, B_ / 16), blk, 0, stream>>>(
            A2, HIDDEN_ + DETER_, HIDDEN_ + DETER_, WpGru, 3 * DETER_, nullptr,
            gru_out, nullptr, nullptr, 0, 0, t);
        // LayerNorm + gates -> deter (carry, output, A2 tail, A4 head)
        gru_ln_gate<<<B_, 256, 0, stream>>>(gru_out, ln_scale, ln_bias,
                                            deterC, out_deter, A2, A4, t);
        // h = elu(deter @ img_out_w + b) -> Hb
        gemm_wmma<0, 2><<<dim3(HIDDEN_ / 128, B_ / 16), blk, 0, stream>>>(
            A4, DETER_ + EMBED_, DETER_, WpImgOut, HIDDEN_, img_out_b,
            nullptr, nullptr, Hb, HIDDEN_, 0, t);
        // prior: h @ img_dist_w + b -> pm / ps (fused transform)
        gemm_wmma<2, 1><<<dim3(2 * STOCH_ / 64, B_ / 16), blk, 0, stream>>>(
            Hb, HIDDEN_, HIDDEN_, WpImgDist, 2 * STOCH_, img_dist_b,
            out_pm, out_ps, nullptr, 0, 0, t);
        // h2 = elu([deter,emb] @ obs_out_w + b) -> H2b
        gemm_wmma<0, 2><<<dim3(HIDDEN_ / 128, B_ / 16), blk, 0, stream>>>(
            A4, DETER_ + EMBED_, DETER_ + EMBED_, WpObsOut, HIDDEN_, obs_out_b,
            nullptr, nullptr, H2b, HIDDEN_, 0, t);
        // posterior logits: h2 @ obs_dist_w + b -> q_raw
        gemm_wmma<3, 1><<<dim3(2 * STOCH_ / 64, B_ / 16), blk, 0, stream>>>(
            H2b, HIDDEN_, HIDDEN_, WpObsDist, 2 * STOCH_, obs_dist_b,
            q_raw, nullptr, nullptr, 0, 0, t);
        // stoch = qm + qs*noise; write outputs + A1 for t+1
        finish_post<<<(unsigned)(((size_t)B_ * STOCH_ + 255) / 256), 256, 0, stream>>>(
            q_raw, noise, out_stoch, out_qm, out_qs, A1, t);
    }
}
